// BrainGPTv2_7876970021237
// MI455X (gfx1250) — matrix operand order repaired and checked by reading, not compiled, not run on hardware
//
#include <hip/hip_runtime.h>
#include <math.h>

// ---------------- problem constants ----------------
#define B_      2
#define L_      2048
#define D_      1024
#define NH_     16
#define HD_     64
#define DI_     2048
#define DC_     4
#define KSEL    204          // int(2048*0.1)
#define MAXSTEP 2
#define THRESH_ 0.99f
#define RL      (B_*L_)      // 4096 rows
#define ND      (RL*D_)      // 4194304
#define NDI     (RL*DI_)     // 8388608

typedef __attribute__((ext_vector_type(16))) _Float16 v16h;
typedef __attribute__((ext_vector_type(8)))  _Float16 v8h;
typedef __attribute__((ext_vector_type(8)))  float    v8f;

__device__ __forceinline__ float wred_sum(float v){
  #pragma unroll
  for (int o=16;o>0;o>>=1) v += __shfl_xor(v,o,32);
  return v;
}
__device__ __forceinline__ float wred_max(float v){
  #pragma unroll
  for (int o=16;o>0;o>>=1) v = fmaxf(v,__shfl_xor(v,o,32));
  return v;
}

// ---------------- generic utility kernels ----------------
__global__ void k_fill(float* __restrict__ p, float v, int n){
  int i = blockIdx.x*blockDim.x + threadIdx.x;
  if (i < n) p[i] = v;
}
__global__ void k_f16cvt(const float* __restrict__ s, _Float16* __restrict__ d, int n){
  int i = blockIdx.x*blockDim.x + threadIdx.x;
  if (i < n) d[i] = (_Float16)s[i];
}
__global__ void k_add(float* __restrict__ dst, const float* __restrict__ src, int n){
  int i = blockIdx.x*blockDim.x + threadIdx.x;
  if (i < n) dst[i] += src[i];
}

// ---------------- WMMA GEMM: out[M,N] = epi(A[M,K] @ W[N,K]^T + bias) (+resid) ----------------
// A, W in f16 (K-contiguous).  epi: 0=none 1=relu 2=gelu(exact)
__global__ __launch_bounds__(256) void k_gemm(
    const _Float16* __restrict__ A, const _Float16* __restrict__ W,
    const float* __restrict__ bias, const float* __restrict__ resid,
    float* __restrict__ outF, _Float16* __restrict__ outH,
    int M, int N, int Kd, int epi)
{
  const int lane = threadIdx.x & 31;
  const int wv   = threadIdx.x >> 5;
  const int rowBase = blockIdx.y * 128 + wv * 16;
  const int colBase = blockIdx.x * 64;
  const int mi = lane & 15;
  const int hf = lane >> 4;

  int ar = rowBase + mi; if (ar > M-1) ar = M-1;
  const _Float16* pA = A + (size_t)ar * Kd + hf * 16;

  int br0 = colBase +  0 + mi; if (br0 > N-1) br0 = N-1;
  int br1 = colBase + 16 + mi; if (br1 > N-1) br1 = N-1;
  int br2 = colBase + 32 + mi; if (br2 > N-1) br2 = N-1;
  int br3 = colBase + 48 + mi; if (br3 > N-1) br3 = N-1;
  const _Float16* pB0 = W + (size_t)br0 * Kd + hf * 16;
  const _Float16* pB1 = W + (size_t)br1 * Kd + hf * 16;
  const _Float16* pB2 = W + (size_t)br2 * Kd + hf * 16;
  const _Float16* pB3 = W + (size_t)br3 * Kd + hf * 16;

  v8f acc0 = {}, acc1 = {}, acc2 = {}, acc3 = {};

  for (int k0 = 0; k0 < Kd; k0 += 32) {
    __builtin_prefetch(pA  + k0 + 128, 0, 0);
    __builtin_prefetch(pB0 + k0 + 128, 0, 0);
    union { v16h v; v8h h[2]; } a;
    a.h[0] = *(const v8h*)(pA + k0);        // K = k0 + hf*16 + [0..7]   -> VGPR0..3
    a.h[1] = *(const v8h*)(pA + k0 + 8);    // K = k0 + hf*16 + [8..15]  -> VGPR4..7
    v16h b0 = *(const v16h*)(pB0 + k0);     // K = k0 + hf*16 + [0..15]
    v16h b1 = *(const v16h*)(pB1 + k0);
    v16h b2 = *(const v16h*)(pB2 + k0);
    v16h b3 = *(const v16h*)(pB3 + k0);
    acc0 = __builtin_amdgcn_wmma_f32_16x16x32_f16(false, a.v, false, b0, (short)0, acc0, false, false);
    acc1 = __builtin_amdgcn_wmma_f32_16x16x32_f16(false, a.v, false, b1, (short)0, acc1, false, false);
    acc2 = __builtin_amdgcn_wmma_f32_16x16x32_f16(false, a.v, false, b2, (short)0, acc2, false, false);
    acc3 = __builtin_amdgcn_wmma_f32_16x16x32_f16(false, a.v, false, b3, (short)0, acc3, false, false);
  }

  // C/D layout: lanes0-15 -> M = hf*8? no: lane<16 => rows rowBase+0..7 (VGPR r -> M=r);
  // lane>=16 => rows rowBase+8..15.  col = colBase + j*16 + (lane&15).
  const int orow0 = rowBase + hf * 8;
  #pragma unroll
  for (int j = 0; j < 4; ++j) {
    v8f acc = (j==0) ? acc0 : (j==1) ? acc1 : (j==2) ? acc2 : acc3;
    int col = colBase + j*16 + mi;
    if (col >= N) continue;
    float bv = bias ? bias[col] : 0.0f;
    #pragma unroll
    for (int r = 0; r < 8; ++r) {
      int row = orow0 + r;
      if (row >= M) continue;
      float v = acc[r] + bv;
      if (epi == 1)      v = v > 0.0f ? v : 0.0f;
      else if (epi == 2) v = 0.5f * v * (1.0f + erff(v * 0.70710678118f));
      size_t off = (size_t)row * N + col;
      if (resid) v += resid[off];
      if (outF)  outF[off] = v;
      if (outH)  outH[off] = (_Float16)v;
    }
  }
}

// ---------------- LayerNorm: one wave per row, writes f16 (+ optional f32) ----------------
__global__ __launch_bounds__(256) void k_ln(
    const float* __restrict__ x, const float* __restrict__ g, const float* __restrict__ bt,
    _Float16* __restrict__ outH, float* __restrict__ outF, int rows, int Dd)
{
  int wv = threadIdx.x >> 5, lane = threadIdx.x & 31;
  int row = blockIdx.x * 8 + wv;
  if (row >= rows) return;
  const float* px = x + (size_t)row * Dd;
  float s = 0.f, s2 = 0.f;
  for (int c = lane; c < Dd; c += 32) { float v = px[c]; s += v; s2 += v*v; }
  s = wred_sum(s); s2 = wred_sum(s2);
  float mean = s / Dd;
  float var  = s2 / Dd - mean*mean;
  float rs   = rsqrtf(var + 1e-5f);
  for (int c = lane; c < Dd; c += 32) {
    float v = (px[c] - mean) * rs * g[c] + bt[c];
    outH[(size_t)row*Dd + c] = (_Float16)v;
    if (outF) outF[(size_t)row*Dd + c] = v;
  }
}

// ---------------- row dot (f16 or f32 src): out[row] = maybe_sigmoid(dot + bias) ----------------
template <typename T>
__global__ __launch_bounds__(256) void k_rowdot(
    const T* __restrict__ src, const float* __restrict__ w, const float* __restrict__ bias,
    float* __restrict__ out, int rows, int Kd, int do_sig)
{
  int wv = threadIdx.x >> 5, lane = threadIdx.x & 31;
  int row = blockIdx.x * 8 + wv;
  if (row >= rows) return;
  const T* p = src + (size_t)row * Kd;
  float s = 0.f;
  for (int c = lane; c < Kd; c += 32) s += (float)p[c] * w[c];
  s = wred_sum(s);
  if (lane == 0) {
    if (bias) s += bias[0];
    if (do_sig) s = 1.0f / (1.0f + expf(-s));
    out[row] = s;
  }
}

// ---------------- exact top-K via full bitonic sort (L=2048) in LDS, one block per batch ----------------
__device__ __forceinline__ bool kv_before(float va, int ia, float vb, int ib){
  return (va > vb) || (va == vb && ia < ib);   // descending value, tie -> lower index first
}
__global__ __launch_bounds__(1024) void k_topk(const float* __restrict__ score, int* __restrict__ idx)
{
  __shared__ float sv[L_];
  __shared__ int   si[L_];
  const int bb = blockIdx.x;
  for (int i = threadIdx.x; i < L_; i += 1024) { sv[i] = score[bb*L_ + i]; si[i] = i; }
  __syncthreads();
  for (int k = 2; k <= L_; k <<= 1) {
    for (int j = k >> 1; j > 0; j >>= 1) {
      for (int i = threadIdx.x; i < L_; i += 1024) {
        int ixj = i ^ j;
        if (ixj > i) {
          bool asc = ((i & k) == 0);
          float vi = sv[i], vx = sv[ixj];
          int   ii = si[i], ix = si[ixj];
          bool sw = asc ? kv_before(vx, ix, vi, ii) : kv_before(vi, ii, vx, ix);
          if (sw) { sv[i]=vx; sv[ixj]=vi; si[i]=ix; si[ixj]=ii; }
        }
      }
      __syncthreads();
    }
  }
  for (int t = threadIdx.x; t < KSEL; t += 1024) idx[bb*KSEL + t] = si[t];
}

// ---------------- gather selected tokens (f16) ----------------
__global__ void k_gather(const _Float16* __restrict__ lnh, const int* __restrict__ idx,
                         _Float16* __restrict__ tok)
{
  int r = blockIdx.x;               // 0 .. B*KSEL-1
  int b = r / KSEL, j = r % KSEL;
  int t = idx[b*KSEL + j];
  const _Float16* src = lnh + ((size_t)b*L_ + t) * D_;
  _Float16* dst = tok + (size_t)r * D_;
  for (int c = threadIdx.x; c < D_; c += blockDim.x) dst[c] = src[c];
}

// ---------------- attention over the K=204 selected tokens, one wave per (q) ----------------
__global__ __launch_bounds__(256) void k_attn(const float* __restrict__ qkv, _Float16* __restrict__ outh)
{
  __shared__ float att[8][KSEL];
  int bh = blockIdx.x; int b = bh / NH_; int h = bh % NH_;
  int wv = threadIdx.x >> 5, lane = threadIdx.x & 31;
  for (int q = wv; q < KSEL; q += 8) {
    const float* qp = qkv + (size_t)(b*KSEL + q)*(3*D_) + h*HD_;
    float q0 = qp[lane], q1 = qp[lane + 32];
    for (int k = 0; k < KSEL; ++k) {
      const float* kp = qkv + (size_t)(b*KSEL + k)*(3*D_) + D_ + h*HD_;
      float part = q0*kp[lane] + q1*kp[lane + 32];
      part = wred_sum(part);
      if (lane == 0) att[wv][k] = part * 0.125f;   // 1/sqrt(64)
    }
    float mx = -3.0e38f;
    for (int k = lane; k < KSEL; k += 32) mx = fmaxf(mx, att[wv][k]);
    mx = wred_max(mx);
    float sum = 0.f;
    for (int k = lane; k < KSEL; k += 32) { float e = expf(att[wv][k] - mx); att[wv][k] = e; sum += e; }
    sum = wred_sum(sum);
    float inv = 1.0f / sum;
    float o0 = 0.f, o1 = 0.f;
    for (int k = 0; k < KSEL; ++k) {
      const float* vp = qkv + (size_t)(b*KSEL + k)*(3*D_) + 2*D_ + h*HD_;
      float a = att[wv][k] * inv;
      o0 += a * vp[lane]; o1 += a * vp[lane + 32];
    }
    _Float16* op = outh + (size_t)(b*KSEL + q)*D_ + h*HD_;
    op[lane] = (_Float16)o0; op[lane + 32] = (_Float16)o1;
  }
}

// ---------------- scatter-add: x[b, idx[b,j], :] += res_w * proj[b*K+j, :] ----------------
__global__ void k_scatter(float* __restrict__ x, const float* __restrict__ proj,
                          const int* __restrict__ idx, const float* __restrict__ res_w)
{
  int r = blockIdx.x;
  int b = r / KSEL, j = r % KSEL;
  int t = idx[b*KSEL + j];
  float rw = res_w[0];
  float* dst = x + ((size_t)b*L_ + t) * D_;
  const float* src = proj + (size_t)r * D_;
  for (int c = threadIdx.x; c < D_; c += blockDim.x) dst[c] += rw * src[c];
}

// ---------------- depthwise causal conv (DC=4) + silu ----------------
__global__ void k_conv(const float* __restrict__ xz, const float* __restrict__ cw,
                       float* __restrict__ xi)
{
  int tid = blockIdx.x*blockDim.x + threadIdx.x;
  if (tid >= NDI) return;
  int c = tid % DI_;
  int t = (tid / DI_) % L_;
  int b = tid / (DI_ * L_);
  const float* col = xz + (size_t)(b * L_) * (2*DI_) + c;
  float s = 0.f;
  #pragma unroll
  for (int j = 0; j < DC_; ++j) {
    int tt = t - (DC_ - 1) + j;
    if (tt >= 0) s += col[(size_t)tt * (2*DI_)] * cw[c*DC_ + j];
  }
  float sig = 1.0f / (1.0f + expf(-s));
  xi[tid] = s * sig;
}

// ---------------- mamba elementwise: y = (xi*sigmoid(dt) + xi*D) * silu(z), f16 out ----------------
__global__ void k_mamba_y(const float* __restrict__ xz, const float* __restrict__ xi,
    const float* __restrict__ sdot, const float* __restrict__ dt_w, const float* __restrict__ dt_b,
    const float* __restrict__ Dp, _Float16* __restrict__ yh)
{
  int tid = blockIdx.x*blockDim.x + threadIdx.x;
  if (tid >= NDI) return;
  int c   = tid % DI_;
  int row = tid / DI_;
  float z   = xz[(size_t)row * (2*DI_) + DI_ + c];
  float xv  = xi[tid];
  float din = sdot[row] * dt_w[c] + dt_b[c];
  float dt  = (din > 20.0f) ? din : log1pf(expf(din));   // softplus
  float sg  = 1.0f / (1.0f + expf(-dt));
  float y   = xv * sg + xv * Dp[c];
  float sz  = z / (1.0f + expf(-z));
  yh[tid] = (_Float16)(y * sz);
}

// ---------------- ACT halting accumulation (one block per row) ----------------
__global__ __launch_bounds__(256) void k_act(const float* __restrict__ x, const float* __restrict__ p,
    float* __restrict__ halting, float* __restrict__ rem, float* __restrict__ out, int use_p)
{
  int row = blockIdx.x;
  float coef;
  if (use_p) {
    float h = halting[row], r = rem[row], pp = p[row];
    float still = (h < THRESH_) ? 1.0f : 0.0f;
    coef = still * fminf(r, pp);
  } else {
    coef = rem[row];
  }
  __syncthreads();
  const float* px = x + (size_t)row * D_;
  float* po = out + (size_t)row * D_;
  for (int c = threadIdx.x; c < D_; c += 256) po[c] += coef * px[c];
  if (use_p && threadIdx.x == 0) { halting[row] += coef; rem[row] -= coef; }
}

// =======================================================================================
extern "C" void kernel_launch(void* const* d_in, const int* in_sizes, int n_in,
                              void* d_out, int out_size, void* d_ws, size_t ws_size,
                              hipStream_t stream)
{
  // ---- inputs in setup_inputs() order ----
  const float* X0       = (const float*)d_in[0];
  const float* hp_w1    = (const float*)d_in[1];
  const float* hp_b1    = (const float*)d_in[2];
  const float* hp_w2    = (const float*)d_in[3];
  const float* hp_b2    = (const float*)d_in[4];
  const float* b0_ln1_g = (const float*)d_in[5];
  const float* b0_ln1_b = (const float*)d_in[6];
  const float* b0_ln2_g = (const float*)d_in[7];
  const float* b0_ln2_b = (const float*)d_in[8];
  const float* imp_w1   = (const float*)d_in[9];
  const float* imp_b1   = (const float*)d_in[10];
  const float* imp_w2   = (const float*)d_in[11];
  const float* imp_b2   = (const float*)d_in[12];
  const float* qkv_w    = (const float*)d_in[13];
  const float* aout_w   = (const float*)d_in[14];
  const float* res_w    = (const float*)d_in[15];
  const float* b0m1     = (const float*)d_in[16];
  const float* b0m2     = (const float*)d_in[17];
  const float* b1_ln1_g = (const float*)d_in[18];
  const float* b1_ln1_b = (const float*)d_in[19];
  const float* b1_ln2_g = (const float*)d_in[20];
  const float* b1_ln2_b = (const float*)d_in[21];
  const float* in_w     = (const float*)d_in[22];
  const float* conv_w   = (const float*)d_in[23];
  const float* xproj_w  = (const float*)d_in[24];
  const float* dt_w     = (const float*)d_in[25];
  const float* dt_b     = (const float*)d_in[26];
  const float* Dp       = (const float*)d_in[27];
  const float* b1out_w  = (const float*)d_in[28];
  const float* b1m1     = (const float*)d_in[29];
  const float* b1m2     = (const float*)d_in[30];

  // ---- workspace carve-up (256B aligned) ----
  char* base = (char*)d_ws;
  size_t cur = 0;
  auto AL = [&](size_t bytes) -> void* {
    void* p = base + cur;
    cur = (cur + bytes + 255) & ~(size_t)255;
    return p;
  };
  _Float16* hp_w1h   = (_Float16*)AL((size_t)256*D_      * 2);
  _Float16* imp_w1h  = (_Float16*)AL((size_t)256*D_      * 2);
  _Float16* qkv_wh   = (_Float16*)AL((size_t)3*D_*D_     * 2);
  _Float16* aout_wh  = (_Float16*)AL((size_t)D_*D_       * 2);
  _Float16* b0m1h    = (_Float16*)AL((size_t)4*D_*D_     * 2);
  _Float16* b0m2h    = (_Float16*)AL((size_t)4*D_*D_     * 2);
  _Float16* in_wh    = (_Float16*)AL((size_t)2*DI_*D_    * 2);
  _Float16* b1out_wh = (_Float16*)AL((size_t)D_*DI_      * 2);
  _Float16* b1m1h    = (_Float16*)AL((size_t)4*D_*D_     * 2);
  _Float16* b1m2h    = (_Float16*)AL((size_t)4*D_*D_     * 2);

  float*    xbuf     = (float*)   AL((size_t)ND * 4);
  _Float16* lnh      = (_Float16*)AL((size_t)ND * 2);
  float*    lnf      = (float*)   AL((size_t)ND * 4);
  _Float16* imp_hidh = (_Float16*)AL((size_t)RL*256 * 2);
  float*    score    = (float*)   AL((size_t)RL * 4);
  int*      idxbuf   = (int*)     AL((size_t)B_*KSEL * 4);
  _Float16* tok_h    = (_Float16*)AL((size_t)B_*KSEL*D_ * 2);
  float*    qkvf     = (float*)   AL((size_t)B_*KSEL*3*D_ * 4);
  _Float16* attout_h = (_Float16*)AL((size_t)B_*KSEL*D_ * 2);
  float*    projf    = (float*)   AL((size_t)B_*KSEL*D_ * 4);
  _Float16* mid_h    = (_Float16*)AL((size_t)RL*4*D_ * 2);
  float*    xzf      = (float*)   AL((size_t)RL*2*DI_ * 4);
  float*    xif      = (float*)   AL((size_t)NDI * 4);
  float*    svec     = (float*)   AL((size_t)RL * 4);
  _Float16* yh       = (_Float16*)AL((size_t)NDI * 2);
  _Float16* hp_hidh  = (_Float16*)AL((size_t)RL*256 * 2);
  float*    pvec     = (float*)   AL((size_t)RL * 4);
  float*    halting  = (float*)   AL((size_t)RL * 4);
  float*    rem      = (float*)   AL((size_t)RL * 4);
  (void)ws_size; (void)in_sizes; (void)n_in; (void)out_size;

  auto CVT = [&](const float* s, _Float16* d, int n){
    k_f16cvt<<<(n + 255)/256, 256, 0, stream>>>(s, d, n);
  };
  auto GEMM = [&](const _Float16* Aa, const _Float16* Ww, const float* bias, const float* resid,
                  float* oF, _Float16* oH, int M, int N, int Kd, int epi){
    dim3 g((N + 63)/64, (M + 127)/128);
    k_gemm<<<g, 256, 0, stream>>>(Aa, Ww, bias, resid, oF, oH, M, N, Kd, epi);
  };
  auto LN = [&](const float* x, const float* g, const float* b, _Float16* oH, float* oF){
    k_ln<<<(RL + 7)/8, 256, 0, stream>>>(x, g, b, oH, oF, RL, D_);
  };

  // ---- weight conversion f32 -> f16 (done every call; deterministic) ----
  CVT(hp_w1,   hp_w1h,   256*D_);
  CVT(imp_w1,  imp_w1h,  256*D_);
  CVT(qkv_w,   qkv_wh,   3*D_*D_);
  CVT(aout_w,  aout_wh,  D_*D_);
  CVT(b0m1,    b0m1h,    4*D_*D_);
  CVT(b0m2,    b0m2h,    4*D_*D_);
  CVT(in_w,    in_wh,    2*DI_*D_);
  CVT(b1out_w, b1out_wh, D_*DI_);
  CVT(b1m1,    b1m1h,    4*D_*D_);
  CVT(b1m2,    b1m2h,    4*D_*D_);

  // ---- init state ----
  hipMemcpyAsync(xbuf, X0, (size_t)ND*4, hipMemcpyDeviceToDevice, stream);
  k_fill<<<(RL+255)/256, 256, 0, stream>>>(halting, 0.0f, RL);
  k_fill<<<(RL+255)/256, 256, 0, stream>>>(rem,     1.0f, RL);
  k_fill<<<(ND+255)/256, 256, 0, stream>>>((float*)d_out, 0.0f, ND);

  for (int step = 0; step < MAXSTEP; ++step) {
    // ======== block 0: selective attention ========
    LN(xbuf, b0_ln1_g, b0_ln1_b, lnh, lnf);                                    // ln1 -> f16 + f32
    GEMM(lnh, imp_w1h, imp_b1, nullptr, nullptr, imp_hidh, RL, 256, D_, 1);    // relu(t@w1.T+b1)
    k_rowdot<_Float16><<<(RL+7)/8, 256, 0, stream>>>(imp_hidh, imp_w2, imp_b2, score, RL, 256, 1);
    k_topk<<<B_, 1024, 0, stream>>>(score, idxbuf);
    k_gather<<<B_*KSEL, 256, 0, stream>>>(lnh, idxbuf, tok_h);
    GEMM(tok_h, qkv_wh, nullptr, nullptr, qkvf, nullptr, B_*KSEL, 3*D_, D_, 0);
    k_attn<<<B_*NH_, 256, 0, stream>>>(qkvf, attout_h);
    GEMM(attout_h, aout_wh, nullptr, nullptr, projf, nullptr, B_*KSEL, D_, D_, 0);
    k_add<<<(ND+255)/256, 256, 0, stream>>>(xbuf, lnf, ND);                    // x += ln1(x) (mixer residual)
    k_scatter<<<B_*KSEL, 256, 0, stream>>>(xbuf, projf, idxbuf, res_w);        // x += res_w*scat
    // MLP b0
    LN(xbuf, b0_ln2_g, b0_ln2_b, lnh, nullptr);
    GEMM(lnh, b0m1h, nullptr, nullptr, nullptr, mid_h, RL, 4*D_, D_, 2);       // gelu
    GEMM(mid_h, b0m2h, nullptr, xbuf, xbuf, nullptr, RL, D_, 4*D_, 0);         // + residual

    // ======== block 1: mamba ========
    LN(xbuf, b1_ln1_g, b1_ln1_b, lnh, nullptr);
    GEMM(lnh, in_wh, nullptr, nullptr, xzf, nullptr, RL, 2*DI_, D_, 0);        // xz
    k_conv<<<(NDI+255)/256, 256, 0, stream>>>(xzf, conv_w, xif);               // silu(causal dwconv)
    k_rowdot<float><<<(RL+7)/8, 256, 0, stream>>>(xif, xproj_w + 32*DI_, nullptr, svec, RL, DI_, 0);
    k_mamba_y<<<(NDI+255)/256, 256, 0, stream>>>(xzf, xif, svec, dt_w, dt_b, Dp, yh);
    GEMM(yh, b1out_wh, nullptr, xbuf, xbuf, nullptr, RL, D_, DI_, 0);          // + residual
    // MLP b1
    LN(xbuf, b1_ln2_g, b1_ln2_b, lnh, nullptr);
    GEMM(lnh, b1m1h, nullptr, nullptr, nullptr, mid_h, RL, 4*D_, D_, 2);
    GEMM(mid_h, b1m2h, nullptr, xbuf, xbuf, nullptr, RL, D_, 4*D_, 0);

    // ======== ACT halting ========
    k_f16cvt<<<(ND+255)/256, 256, 0, stream>>>(xbuf, lnh, ND);                 // x -> f16 (reuse lnh)
    GEMM(lnh, hp_w1h, hp_b1, nullptr, nullptr, hp_hidh, RL, 256, D_, 1);
    k_rowdot<_Float16><<<(RL+7)/8, 256, 0, stream>>>(hp_hidh, hp_w2, hp_b2, pvec, RL, 256, 1);
    k_act<<<RL, 256, 0, stream>>>(xbuf, pvec, halting, rem, (float*)d_out, 1);
  }
  // out += rem * x
  k_act<<<RL, 256, 0, stream>>>(xbuf, pvec, halting, rem, (float*)d_out, 0);
}
